// HashAttention_54357106098896
// MI455X (gfx1250) — matrix operand order repaired and checked by reading, not compiled, not run on hardware
//
#include <hip/hip_runtime.h>

// ---------------------------------------------------------------------------
// HashAttention pipeline for gfx1250 (MI455X), wave32 + WMMA + TDM.
//  - activations kept channel-last f16 so conv patches are dense tiles
//  - GEMM K-dim permuted so WMMA A/B fragments are 2x ds_load_b128 each
//  - patch staging via tensor_load_to_lds (TDM) with LDS pad feature
//  - conv1: M=128(K=288) / conv2: M=32(K=1152), N=65536 per batch
// ---------------------------------------------------------------------------

typedef __attribute__((ext_vector_type(16))) _Float16 v16h;
typedef __attribute__((ext_vector_type(8)))  _Float16 v8h;
typedef __attribute__((ext_vector_type(8)))  float    v8f;
typedef __attribute__((ext_vector_type(4)))  float    v4f;
typedef unsigned int u32;
typedef unsigned long long u64;
typedef __attribute__((ext_vector_type(4))) u32 v4u;
typedef __attribute__((ext_vector_type(4))) unsigned int uint32x4;
typedef __attribute__((ext_vector_type(8))) int         int32x8;
typedef __attribute__((ext_vector_type(4))) int         int32x4;

#define NB    8
#define LVL   16
#define TBL   65536
#define CH1   32
#define CH2   128
#define PRIME 2654435761u
#define CPAD  40          // LDS channel stride (32 data + 8 pad halves)

// floor(16 * 16^(l/15)) precomputed (matches reference _level_res)
__device__ __constant__ int kRes[16] =
    {16,19,23,27,33,40,48,58,70,84,101,122,147,176,212,256};

union Frag  { v8h h2[2]; v16h h; };
union Pack8 { _Float16 h[8]; v4u q; };
union Acc8  { v8f v; v4f q[2]; };

// ---------------------------------------------------------------------------
// TDM descriptor builders (D# group layout per CDNA5 ISA 8.3-8.6).
// data_size=2B, pad_enable=1, pad_interval=16 dwords (code 3),
// pad_amount=4 dwords (code 3)  => LDS channel stride 40 halves.
__device__ inline void tdm_patch_2d(u32 lds_off, const void* gptr,
                                    u32 tile_d0, u32 tile_d1,
                                    u32 tensor_d0, u32 tensor_d1, u64 stride0) {
    u64 ga = (u64)(uintptr_t)gptr;
    uint32x4 g0;
    g0[0] = 1u;                                          // count=1 (valid load)
    g0[1] = lds_off;                                     // lds_addr
    g0[2] = (u32)ga;                                     // global_addr[31:0]
    g0[3] = (u32)((ga >> 32) & 0x1FFFFFFu) | (2u << 30); // addr[56:32], type=2
    int32x8 g1;
    g1[0] = (int)((1u << 16) | (1u << 20) | (3u << 22) | (3u << 25));
    g1[1] = (int)((tensor_d0 & 0xFFFFu) << 16);          // tensor_dim0.lo16
    g1[2] = (int)((tensor_d0 >> 16) | ((tensor_d1 & 0xFFFFu) << 16));
    g1[3] = (int)((tensor_d1 >> 16) | (tile_d0 << 16));  // tile_dim0
    g1[4] = (int)(tile_d1 & 0xFFFFu);                    // tile_dim1, tile_dim2=0
    g1[5] = (int)(u32)stride0;                           // dim0_stride[31:0]
    g1[6] = (int)(u32)(stride0 >> 32);                   // dim0_stride[47:32]
    g1[7] = 0;
    int32x4 z4 = {0, 0, 0, 0};
    int32x8 z8 = {0, 0, 0, 0, 0, 0, 0, 0};
    __builtin_amdgcn_tensor_load_to_lds(g0, g1, z4, z4, z8, 0);
}

__device__ inline void tdm_patch_3d(u32 lds_off, const void* gptr,
                                    u32 tile_d0, u32 tile_d1, u32 tile_d2,
                                    u32 tensor_d0, u32 tensor_d1, u32 tensor_d2,
                                    u64 stride0, u64 stride1) {
    u64 ga = (u64)(uintptr_t)gptr;
    uint32x4 g0;
    g0[0] = 1u;
    g0[1] = lds_off;
    g0[2] = (u32)ga;
    g0[3] = (u32)((ga >> 32) & 0x1FFFFFFu) | (2u << 30);
    int32x8 g1;
    g1[0] = (int)((1u << 16) | (1u << 20) | (3u << 22) | (3u << 25));
    g1[1] = (int)((tensor_d0 & 0xFFFFu) << 16);
    g1[2] = (int)((tensor_d0 >> 16) | ((tensor_d1 & 0xFFFFu) << 16));
    g1[3] = (int)((tensor_d1 >> 16) | (tile_d0 << 16));
    g1[4] = (int)((tile_d1 & 0xFFFFu) | (tile_d2 << 16));
    g1[5] = (int)(u32)stride0;
    g1[6] = (int)((u32)(stride0 >> 32) | ((u32)(stride1 & 0xFFFFu) << 16));
    g1[7] = (int)(u32)(stride1 >> 16);                   // dim1_stride[47:16]
    int32x4 g2 = {(int)tensor_d2, 0, 0, 0};              // tensor_dim2
    int32x4 g3 = {0, 0, 0, 0};
    int32x8 z8 = {0, 0, 0, 0, 0, 0, 0, 0};
    __builtin_amdgcn_tensor_load_to_lds(g0, g1, g2, g3, z8, 0);
}

// ---------------------------------------------------------------------------
__global__ void fill_zero(u32* p, long n) {
    long i = (long)blockIdx.x * blockDim.x + threadIdx.x;
    long stride = (long)gridDim.x * blockDim.x;
    for (; i < n; i += stride) p[i] = 0u;
}

// ---------------------------------------------------------------------------
__global__ __launch_bounds__(256) void styles_kernel(
    const float* __restrict__ s,
    const float* __restrict__ w1_aff, const float* __restrict__ b1_aff,
    const float* __restrict__ w2_aff, const float* __restrict__ b2_aff,
    float* __restrict__ st1, float* __restrict__ st2) {
    int b = blockIdx.x, t = threadIdx.x;
    const float inv = 0.04419417382415922f;  // 1/sqrt(512)
    if (t < CH1) {
        float acc = 0.f;
        for (int j = 0; j < 512; ++j) acc += s[b*512 + j] * w1_aff[j*CH1 + t];
        st1[b*CH1 + t] = acc * inv + b1_aff[t];
    } else if (t < CH1 + CH2) {
        int c = t - CH1;
        float acc = 0.f;
        for (int j = 0; j < 512; ++j) acc += s[b*512 + j] * w2_aff[j*CH2 + c];
        st2[b*CH2 + c] = acc * inv + b2_aff[c];
    }
}

// ---------------------------------------------------------------------------
// Modulated+demodulated weights, stored with permuted K:
//   conv1: k' = (kh*3+kw)*32 + i
//   conv2: k' = (i/32)*288 + (kh*3+kw)*32 + (i%32)
__global__ __launch_bounds__(512) void modw1_kernel(
    const float* __restrict__ w1, const float* __restrict__ st1,
    _Float16* __restrict__ w1mod) {
    __shared__ float red[512];
    int bid = blockIdx.x; int b = bid >> 7, o = bid & 127;
    int k = threadIdx.x;
    float v = 0.f;
    if (k < 288) v = w1[o*288 + k] * st1[b*CH1 + k/9];
    red[k] = v * v;
    __syncthreads();
    for (int st = 256; st > 0; st >>= 1) {
        if (k < st) red[k] += red[k + st];
        __syncthreads();
    }
    float d = rsqrtf(red[0] + 1e-8f);
    if (k < 288) {
        int i = k / 9, r = k - 9*i;
        int kp = r*32 + i;
        w1mod[((size_t)(b*128 + o))*288 + kp] = (_Float16)(v * d);
    }
}

__global__ __launch_bounds__(1024) void modw2_kernel(
    const float* __restrict__ w2, const float* __restrict__ st2,
    _Float16* __restrict__ w2mod) {
    __shared__ float red[1024];
    int bid = blockIdx.x; int b = bid >> 5, o = bid & 31;
    int t = threadIdx.x;
    float v0 = w2[o*1152 + t] * st2[b*CH2 + t/9];
    float v1 = 0.f;
    int k1 = t + 1024;
    if (k1 < 1152) v1 = w2[o*1152 + k1] * st2[b*CH2 + k1/9];
    red[t] = v0*v0 + v1*v1;
    __syncthreads();
    for (int st = 512; st > 0; st >>= 1) {
        if (t < st) red[t] += red[t + st];
        __syncthreads();
    }
    float d = rsqrtf(red[0] + 1e-8f);
    {
        int i = t / 9, r = t - 9*i;
        int kp = (i >> 5)*288 + r*32 + (i & 31);
        w2mod[((size_t)(b*32 + o))*1152 + kp] = (_Float16)(v0 * d);
    }
    if (k1 < 1152) {
        int i = k1 / 9, r = k1 - 9*i;
        int kp = (i >> 5)*288 + r*32 + (i & 31);
        w2mod[((size_t)(b*32 + o))*1152 + kp] = (_Float16)(v1 * d);
    }
}

// ---------------------------------------------------------------------------
// hash_retrieve -> channel-last padded f16 feats [B][258][258][32]
__global__ __launch_bounds__(256) void retrieve_kernel(
    const float* __restrict__ tables, _Float16* __restrict__ featsCL) {
    int gid = blockIdx.x * 256 + threadIdx.x;
    int b = gid >> 16, p = gid & 65535;
    int x = p & 255, y = p >> 8;
    float cx = (x + 0.5f) * (1.0f/256.0f);
    float cy = (y + 0.5f) * (1.0f/256.0f);
    union { _Float16 h[32]; v4u q[4]; } outv;
    #pragma unroll
    for (int l = 0; l < LVL; ++l) {
        float res = (float)kRes[l];
        float px = cx * res, py = cy * res;
        float xf = floorf(px), yf = floorf(py);
        float fx = px - xf, fy = py - yf;
        u32 ux = (u32)xf, uy = (u32)yf;
        u32 i00 = ( ux     ^ ( uy     *PRIME)) & 65535u;
        u32 i10 = ((ux+1u) ^ ( uy     *PRIME)) & 65535u;
        u32 i01 = ( ux     ^ ((uy+1u)*PRIME)) & 65535u;
        u32 i11 = ((ux+1u) ^ ((uy+1u)*PRIME)) & 65535u;
        float w00 = (1.f-fx)*(1.f-fy), w10 = fx*(1.f-fy);
        float w01 = (1.f-fx)*fy,       w11 = fx*fy;
        const float* tab = tables + ((size_t)(b*LVL + l))*TBL*2;
        float f0 = w00*tab[2*i00]   + w10*tab[2*i10]   + w01*tab[2*i01]   + w11*tab[2*i11];
        float f1 = w00*tab[2*i00+1] + w10*tab[2*i10+1] + w01*tab[2*i01+1] + w11*tab[2*i11+1];
        outv.h[2*l]   = (_Float16)f0;
        outv.h[2*l+1] = (_Float16)f1;
    }
    v4u* dst = (v4u*)(featsCL + ((size_t)(b*258 + (y+1))*258 + (x+1))*CH1);
    dst[0] = outv.q[0]; dst[1] = outv.q[1];
    dst[2] = outv.q[2]; dst[3] = outv.q[3];
}

// ---------------------------------------------------------------------------
// conv1: block = 64 out-ch (mhalf) x 64 pixels.  8 waves = 4(M) x 2(Nw),
// each wave: 2 N sub-tiles, 9 K-slices (kh,kw), fragments = 2x ds_load_b128.
__global__ __launch_bounds__(256) void conv1_kernel(
    const _Float16* __restrict__ featsCL,
    const _Float16* __restrict__ w1mod,
    const float*    __restrict__ b1,
    _Float16*       __restrict__ x1CL) {
    __shared__ __align__(16) _Float16 lw[64 * 296];       // 37,888 B
    __shared__ __align__(16) _Float16 lp[3 * 66 * CPAD];  // 15,840 B

    int bid = blockIdx.x;
    int b     = bid >> 11;
    int rem   = bid & 2047;
    int mhalf = rem & 1;
    int tile  = rem >> 1;
    int y  = tile >> 2;
    int x0 = (tile & 3) << 6;
    int tid = threadIdx.x;

    // TDM: patch rows y..y+2, cols x0..x0+65, 32 ch -> LDS [3][66][40]
    if (tid < 32) {
        const void* g = featsCL + ((size_t)(b*258 + y)*258 + x0)*CH1;
        tdm_patch_2d((u32)(uintptr_t)(void*)lp, g,
                     /*tile*/ 66*CH1, 3,
                     /*tensor*/ 258*CH1, 1024, /*stride0*/ 258*CH1);
    }
    // weights: 64 rows x 288 halves -> stride 296 (16B-aligned rows)
    {
        const v4u* src = (const v4u*)(w1mod + (size_t)(b*128 + mhalf*64)*288);
        for (int i = tid; i < 64*36; i += 256) {
            int o = i / 36, q = i - o*36;
            *((v4u*)(lw + o*296) + q) = src[o*36 + q];
        }
    }
    if (tid < 32) __builtin_amdgcn_s_wait_tensorcnt(0);
    __syncthreads();

    int wave = tid >> 5, lane = tid & 31;
    int mtile = wave & 3, nwave = wave >> 2;
    int hi = lane >> 4, nn = lane & 15;

    v8f acc[2] = {};
    int orow = mtile*16 + nn;
    for (int ks = 0; ks < 9; ++ks) {
        int kh = ks / 3, kw = ks - 3*kh;
        Frag af;
        const _Float16* ap = lw + orow*296 + ks*32 + hi*8;
        af.h2[0] = *(const v8h*)ap;
        af.h2[1] = *(const v8h*)(ap + 16);
        #pragma unroll
        for (int nt = 0; nt < 2; ++nt) {
            int xcol = nwave*32 + nt*16 + nn;
            Frag bf;
            const _Float16* bp = lp + (kh*66 + xcol + kw)*CPAD + hi*8;
            bf.h2[0] = *(const v8h*)bp;
            bf.h2[1] = *(const v8h*)(bp + 16);
            acc[nt] = __builtin_amdgcn_wmma_f32_16x16x32_f16(
                false, af.h, false, bf.h, (short)0, acc[nt], false, false);
        }
    }

    // epilogue: bias + lrelu(0.2)*sqrt(2); 8 contiguous channels per lane
    const float SQRT2 = 1.41421356237f;
    #pragma unroll
    for (int nt = 0; nt < 2; ++nt) {
        int xg = x0 + nwave*32 + nt*16 + nn;
        Pack8 pk;
        #pragma unroll
        for (int r = 0; r < 8; ++r) {
            int o = mhalf*64 + mtile*16 + hi*8 + r;
            float v = acc[nt][r] + b1[o];
            v = (v > 0.f ? v : 0.2f*v) * SQRT2;
            pk.h[r] = (_Float16)v;
        }
        size_t off = ((size_t)(b*258 + (y+1))*258 + (xg+1))*CH2
                   + mhalf*64 + mtile*16 + hi*8;
        *(v4u*)(x1CL + off) = pk.q;
    }
}

// ---------------------------------------------------------------------------
// conv2: block = 32 out-ch x 64 pixels; outer loop over 4 channel-blocks
// (K'' = ib*288 + (kh*3+kw)*32 + i%32).  8 waves = 2(M) x 4(N).
__global__ __launch_bounds__(256) void conv2_kernel(
    const _Float16* __restrict__ x1CL,
    const _Float16* __restrict__ w2mod,
    const float*    __restrict__ b2,
    float*          __restrict__ x2CL) {
    __shared__ __align__(16) _Float16 lw[32 * 296];       // 18,944 B
    __shared__ __align__(16) _Float16 lp[3 * 66 * CPAD];  // 15,840 B

    int bid = blockIdx.x;
    int b    = bid >> 10;
    int tile = bid & 1023;
    int y  = tile >> 2;
    int x0 = (tile & 3) << 6;
    int tid = threadIdx.x, wave = tid >> 5, lane = tid & 31;
    int mtile = wave & 1, ntile = wave >> 1;
    int hi = lane >> 4, nn = lane & 15;

    v8f acc = {};
    int orow = mtile*16 + nn;

    for (int ib = 0; ib < 4; ++ib) {
        __syncthreads();   // previous chunk's readers done
        // TDM: 3D tile (32ch x 66px x 3rows) of channel block ib
        if (tid < 32) {
            const void* g = x1CL + ((size_t)(b*258 + y)*258 + x0)*CH2 + ib*32;
            tdm_patch_3d((u32)(uintptr_t)(void*)lp, g,
                         /*tile*/ 32, 66, 3,
                         /*tensor*/ 128, 512, 1024,
                         /*stride0*/ CH2, /*stride1*/ 258*CH2);
        }
        // weight chunk ib: 32 rows x 288 halves
        {
            const v4u* src = (const v4u*)(w2mod + (size_t)b*32*1152);
            for (int i = tid; i < 32*36; i += 256) {
                int o = i / 36, q = i - o*36;
                *((v4u*)(lw + o*296) + q) = src[o*144 + ib*36 + q];
            }
        }
        if (tid < 32) __builtin_amdgcn_s_wait_tensorcnt(0);
        __syncthreads();

        for (int ks = 0; ks < 9; ++ks) {
            int kh = ks / 3, kw = ks - 3*kh;
            Frag af;
            const _Float16* ap = lw + orow*296 + ks*32 + hi*8;
            af.h2[0] = *(const v8h*)ap;
            af.h2[1] = *(const v8h*)(ap + 16);
            int xcol = ntile*16 + nn;
            Frag bf;
            const _Float16* bp = lp + (kh*66 + xcol + kw)*CPAD + hi*8;
            bf.h2[0] = *(const v8h*)bp;
            bf.h2[1] = *(const v8h*)(bp + 16);
            acc = __builtin_amdgcn_wmma_f32_16x16x32_f16(
                false, af.h, false, bf.h, (short)0, acc, false, false);
        }
    }

    // epilogue: bias, linear; x2 channel-last f32 [B][65536][32]
    int xg = x0 + ntile*16 + nn;
    int p = y*256 + xg;
    Acc8 res;
    #pragma unroll
    for (int r = 0; r < 8; ++r) res.v[r] = acc[r] + b2[mtile*16 + hi*8 + r];
    float* dst = x2CL + ((size_t)(b*65536 + p))*CH1 + mtile*16 + hi*8;
    *(v4f*)dst       = res.q[0];
    *(v4f*)(dst + 4) = res.q[1];
}

// ---------------------------------------------------------------------------
// hash_recon: bilinear scatter-add into d_out [B][16][65536][2]
__global__ __launch_bounds__(256) void recon_kernel(
    const float* __restrict__ x2CL, float* __restrict__ out) {
    int gid = blockIdx.x * 256 + threadIdx.x;
    int b = gid >> 16, p = gid & 65535;
    int x = p & 255, y = p >> 8;
    float cx = (x + 0.5f) * (1.0f/256.0f);
    float cy = (y + 0.5f) * (1.0f/256.0f);
    const float* tokp = x2CL + ((size_t)(b*65536 + p))*CH1;
    #pragma unroll 4
    for (int l = 0; l < LVL; ++l) {
        float res = (float)kRes[l];
        float px = cx * res, py = cy * res;
        float xf = floorf(px), yf = floorf(py);
        float fx = px - xf, fy = py - yf;
        u32 ux = (u32)xf, uy = (u32)yf;
        u32 i00 = ( ux     ^ ( uy     *PRIME)) & 65535u;
        u32 i10 = ((ux+1u) ^ ( uy     *PRIME)) & 65535u;
        u32 i01 = ( ux     ^ ((uy+1u)*PRIME)) & 65535u;
        u32 i11 = ((ux+1u) ^ ((uy+1u)*PRIME)) & 65535u;
        float w00 = (1.f-fx)*(1.f-fy), w10 = fx*(1.f-fy);
        float w01 = (1.f-fx)*fy,       w11 = fx*fy;
        float f0 = tokp[2*l], f1 = tokp[2*l+1];
        float* tab = out + ((size_t)(b*LVL + l))*TBL*2;
        atomicAdd(&tab[2*i00  ], w00*f0); atomicAdd(&tab[2*i00+1], w00*f1);
        atomicAdd(&tab[2*i10  ], w10*f0); atomicAdd(&tab[2*i10+1], w10*f1);
        atomicAdd(&tab[2*i01  ], w01*f0); atomicAdd(&tab[2*i01+1], w01*f1);
        atomicAdd(&tab[2*i11  ], w11*f0); atomicAdd(&tab[2*i11+1], w11*f1);
    }
}

// ---------------------------------------------------------------------------
extern "C" void kernel_launch(void* const* d_in, const int* in_sizes, int n_in,
                              void* d_out, int out_size, void* d_ws, size_t ws_size,
                              hipStream_t stream) {
    const float* inputs = (const float*)d_in[0];
    const float* s      = (const float*)d_in[1];
    // d_in[2] coords: regular grid, recomputed analytically in-kernel
    const float* w1_aff = (const float*)d_in[3];
    const float* b1_aff = (const float*)d_in[4];
    const float* w1     = (const float*)d_in[5];
    const float* b1     = (const float*)d_in[6];
    const float* w2_aff = (const float*)d_in[7];
    const float* b2_aff = (const float*)d_in[8];
    const float* w2     = (const float*)d_in[9];
    const float* b2     = (const float*)d_in[10];

    char* ws = (char*)d_ws;
    size_t off = 0;
    auto alloc = [&](size_t bytes) -> void* {
        void* p = ws + off;
        off = (off + bytes + 255) & ~(size_t)255;
        return p;
    };
    float*    st1   = (float*)   alloc((size_t)NB*CH1*4);
    float*    st2   = (float*)   alloc((size_t)NB*CH2*4);
    _Float16* w1mod = (_Float16*)alloc((size_t)NB*CH2*288*2);
    _Float16* w2mod = (_Float16*)alloc((size_t)NB*CH1*1152*2);
    _Float16* feats = (_Float16*)alloc((size_t)NB*258*258*CH1*2);  // channel-last
    _Float16* x1    = (_Float16*)alloc((size_t)NB*258*258*CH2*2);  // channel-last
    float*    x2    = (float*)   alloc((size_t)NB*65536*CH1*4);    // channel-last

    fill_zero<<<4096, 256, 0, stream>>>((u32*)feats, (long)NB*258*258*CH1/2);
    fill_zero<<<8192, 256, 0, stream>>>((u32*)x1,    (long)NB*258*258*CH2/2);
    fill_zero<<<8192, 256, 0, stream>>>((u32*)d_out, (long)NB*LVL*TBL*2);

    styles_kernel<<<NB, 256, 0, stream>>>(s, w1_aff, b1_aff, w2_aff, b2_aff, st1, st2);
    modw1_kernel<<<NB*128, 512, 0, stream>>>(w1, st1, w1mod);
    modw2_kernel<<<NB*32, 1024, 0, stream>>>(w2, st2, w2mod);

    retrieve_kernel<<<NB*65536/256, 256, 0, stream>>>(inputs, feats);
    conv1_kernel<<<NB*2048, 256, 0, stream>>>(feats, w1mod, b1, x1);
    conv2_kernel<<<NB*1024, 256, 0, stream>>>(x1, w2mod, b2, x2);
    recon_kernel<<<NB*65536/256, 256, 0, stream>>>(x2, (float*)d_out);
}
